// Model_38113539785432
// MI455X (gfx1250) — matrix operand order repaired and checked by reading, not compiled, not run on hardware
//
#include <hip/hip_runtime.h>
#include <hip/hip_bf16.h>

// ---------------- problem constants (from reference) ----------------
constexpr int Tn   = 2048;          // B*S tokens
constexpr int Hd   = 1024;          // hidden
constexpr int Id   = 3584;          // intermediate
constexpr int En   = 8;             // experts
constexpr int Kt   = 2;             // top-k
constexpr int CAP  = Tn * Kt;       // per-expert worst-case slot capacity (4096)
constexpr int MT   = 32;            // token tile (2 WMMA M-subtiles)

// ---------------- types ----------------
typedef __bf16 bf16x16 __attribute__((ext_vector_type(16)));
typedef __bf16 bf16x2  __attribute__((ext_vector_type(2)));
typedef float  f32x8   __attribute__((ext_vector_type(8)));
typedef float  f32x2   __attribute__((ext_vector_type(2)));

// ---------------- fp32 -> bf16 conversion ----------------
// Express as a real IR fptrunc so the gfx1250 backend can select the hardware
// packed convert (v_cvt_pk_bf16_f32) instead of a software RNE bit sequence.
__device__ __forceinline__ unsigned cvt2_bf16(float a, float b) {
    f32x2 f = { a, b };
    union { bf16x2 v; unsigned u; } c;
    c.v = __builtin_convertvector(f, bf16x2);       // <2 x float> fptrunc <2 x bfloat>
    return c.u;
}
__device__ __forceinline__ unsigned short cvt1_bf16(float a) {
    union { __bf16 h; unsigned short u; } c;
    c.h = (__bf16)a;                                 // scalar fptrunc
    return c.u;
}

// ---------------- WMMA wrapper: D = A(16x32 bf16) * B(32x16 bf16) + C(f32) ----------------
__device__ __forceinline__ f32x8 wmma_bf16(bf16x16 a, bf16x16 b, f32x8 c) {
    return __builtin_amdgcn_wmma_f32_16x16x32_bf16(
        /*neg_a=*/false, a, /*neg_b=*/false, b,
        /*c_mod=*/(short)0, c, /*reuse_a=*/false, /*reuse_b=*/false);
}

// B operand: weight row (= output col n), 16 contiguous fp32 at K = kbase + (lane/16)*16.
// VGPR v holds K = (lane/16)*16 + {2v, 2v+1}  -> contiguous K order, same as the A operand from load_a_lds.
__device__ __forceinline__ bf16x16 load_b_cvt(const float* __restrict__ p) {
    union { bf16x16 v; unsigned u[8]; } r;
    const float4* q = (const float4*)p;
    float4 f0 = q[0], f1 = q[1], f2 = q[2], f3 = q[3];
    r.u[0] = cvt2_bf16(f0.x, f0.y); r.u[1] = cvt2_bf16(f0.z, f0.w);
    r.u[2] = cvt2_bf16(f1.x, f1.y); r.u[3] = cvt2_bf16(f1.z, f1.w);
    r.u[4] = cvt2_bf16(f2.x, f2.y); r.u[5] = cvt2_bf16(f2.z, f2.w);
    r.u[6] = cvt2_bf16(f3.x, f3.y); r.u[7] = cvt2_bf16(f3.z, f3.w);
    return r.v;
}

// A operand from LDS (bf16 row-major, row = lane%16):
// VGPRs 0-7 = K kbase+(lane/16)*16..+15 (contiguous, matches B)
__device__ __forceinline__ bf16x16 load_a_lds(const unsigned short* __restrict__ row,
                                              int kbase, int laneHi) {
    union { bf16x16 v; uint4 q[2]; } r;
    r.q[0] = *(const uint4*)(row + kbase + laneHi * 16);
    r.q[1] = *(const uint4*)(row + kbase + laneHi * 16 + 8);
    return r.v;
}

// ---------------- kernel 0: zero output + expert counters ----------------
__global__ void zero_kernel(float* __restrict__ out, int n, int* __restrict__ counts) {
    int i = blockIdx.x * blockDim.x + threadIdx.x;
    if (i < En) counts[i] = 0;
    for (int j = i; j < n; j += gridDim.x * blockDim.x) out[j] = 0.0f;
}

// ---------------- kernel 1: routing scatter ----------------
__global__ void route_kernel(const int* __restrict__ idx, const float* __restrict__ w,
                             int* __restrict__ counts, int* __restrict__ tok,
                             float* __restrict__ tw) {
    int t = blockIdx.x * blockDim.x + threadIdx.x;
    if (t >= Tn) return;
#pragma unroll
    for (int k = 0; k < Kt; ++k) {
        int e = idx[t * Kt + k];
        int slot = atomicAdd(&counts[e], 1);
        tok[e * CAP + slot] = t;
        tw[e * CAP + slot]  = w[t * Kt + k];
    }
}

// ---------------- kernel 2: fused expert tile (gate/up GEMM + silu + down GEMM) ----------------
// Dynamic LDS layout: sx[MT][Hd] bf16 (64KB) | sh[MT][Id] bf16 (224KB) | stok[MT] | stw[MT]
__global__ void __launch_bounds__(256, 1)
moe_kernel(const float* __restrict__ x,
           const float* __restrict__ gate, const float* __restrict__ up,
           const float* __restrict__ down,
           const int* __restrict__ counts, const int* __restrict__ tok,
           const float* __restrict__ tw, float* __restrict__ out) {
    extern __shared__ char smem_raw[];
    unsigned short* sx   = (unsigned short*)smem_raw;          // [MT][Hd]
    unsigned short* sh   = sx + MT * Hd;                       // [MT][Id]
    int*            stok = (int*)(sh + MT * Id);               // [MT]
    float*          stw  = (float*)(stok + MT);                // [MT]

    const int e    = blockIdx.y;
    const int cnt  = counts[e];
    const int base = blockIdx.x * MT;
    if (base >= cnt) return;                                   // uniform early exit

    const int tid    = threadIdx.x;
    const int lane   = tid & 31;
    const int wave   = tid >> 5;
    const int laneLo = lane & 15;
    const int laneHi = lane >> 4;

    // --- token list for this tile (pad with token 0, weight 0) ---
    if (tid < MT) {
        int s = base + tid;
        int tk = 0; float wt = 0.0f;
        if (s < cnt) { tk = tok[e * CAP + s]; wt = tw[e * CAP + s]; }
        stok[tid] = tk; stw[tid] = wt;
    }
    __syncthreads();

    // --- gather x rows -> LDS as bf16 (coalesced float4 loads) ---
    for (int i = tid; i < MT * (Hd / 4); i += blockDim.x) {
        int r  = i / (Hd / 4);
        int c4 = i % (Hd / 4);
        float4 f = ((const float4*)(x + (size_t)stok[r] * Hd))[c4];
        uint2 pk = { cvt2_bf16(f.x, f.y), cvt2_bf16(f.z, f.w) };
        *(uint2*)(sx + r * Hd + c4 * 4) = pk;
    }
    __syncthreads();

    const float* Wg = gate + (size_t)e * Id * Hd;
    const float* Wu = up   + (size_t)e * Id * Hd;
    const float* Wd = down + (size_t)e * Hd * Id;

    // ================= Stage A: h = silu(x Wg^T) * (x Wu^T), h in LDS =================
    for (int nt = wave; nt < Id / 16; nt += 8) {               // 224/8 = 28 per wave
        const int ibase = nt * 16;
        f32x8 ag0 = {}, ag1 = {}, au0 = {}, au1 = {};
        const float* gRow = Wg + (size_t)(ibase + laneLo) * Hd + laneHi * 16;
        const float* uRow = Wu + (size_t)(ibase + laneLo) * Hd + laneHi * 16;
#pragma unroll 4
        for (int k = 0; k < Hd; k += 32) {
            __builtin_prefetch(gRow + k + 512, 0, 1);          // global_prefetch_b8
            __builtin_prefetch(uRow + k + 512, 0, 1);
            bf16x16 bg = load_b_cvt(gRow + k);
            bf16x16 bu = load_b_cvt(uRow + k);
            bf16x16 a0 = load_a_lds(sx + (size_t)laneLo * Hd, k, laneHi);
            bf16x16 a1 = load_a_lds(sx + (size_t)(16 + laneLo) * Hd, k, laneHi);
            ag0 = wmma_bf16(a0, bg, ag0);
            ag1 = wmma_bf16(a1, bg, ag1);
            au0 = wmma_bf16(a0, bu, au0);
            au1 = wmma_bf16(a1, bu, au1);
        }
        // silu(g)*u in f32, store bf16 h.  C layout: M = laneHi*8 + v, N = laneLo.
#pragma unroll
        for (int v = 0; v < 8; ++v) {
            float g0 = ag0[v], g1 = ag1[v];
            float h0 = (g0 / (1.0f + __expf(-g0))) * au0[v];
            float h1 = (g1 / (1.0f + __expf(-g1))) * au1[v];
            int m0 = laneHi * 8 + v;
            sh[(size_t)m0 * Id + ibase + laneLo]        = cvt1_bf16(h0);
            sh[(size_t)(16 + m0) * Id + ibase + laneLo] = cvt1_bf16(h1);
        }
    }
    __syncthreads();

    // ================= Stage B: out += cw * (h Wd^T) =================
    for (int nt = wave; nt < Hd / 16; nt += 8) {               // 64/8 = 8 per wave
        const int hbase = nt * 16;
        f32x8 ac0 = {}, ac1 = {};
        const float* dRow = Wd + (size_t)(hbase + laneLo) * Id + laneHi * 16;
#pragma unroll 4
        for (int k = 0; k < Id; k += 32) {
            __builtin_prefetch(dRow + k + 512, 0, 1);
            bf16x16 b  = load_b_cvt(dRow + k);
            bf16x16 a0 = load_a_lds(sh + (size_t)laneLo * Id, k, laneHi);
            bf16x16 a1 = load_a_lds(sh + (size_t)(16 + laneLo) * Id, k, laneHi);
            ac0 = wmma_bf16(a0, b, ac0);
            ac1 = wmma_bf16(a1, b, ac1);
        }
#pragma unroll
        for (int v = 0; v < 8; ++v) {
            int m0 = laneHi * 8 + v;
            int m1 = 16 + m0;
            atomicAdd(out + (size_t)stok[m0] * Hd + hbase + laneLo, ac0[v] * stw[m0]);
            atomicAdd(out + (size_t)stok[m1] * Hd + hbase + laneLo, ac1[v] * stw[m1]);
        }
    }
}

// ---------------- host launcher ----------------
extern "C" void kernel_launch(void* const* d_in, const int* in_sizes, int n_in,
                              void* d_out, int out_size, void* d_ws, size_t ws_size,
                              hipStream_t stream) {
    const float* x    = (const float*)d_in[0];
    const int*   eidx = (const int*)d_in[1];
    const float* ew   = (const float*)d_in[2];
    const float* gate = (const float*)d_in[3];
    const float* up   = (const float*)d_in[4];
    const float* down = (const float*)d_in[5];
    float* out = (float*)d_out;

    char* ws = (char*)d_ws;
    int*   counts = (int*)ws;                                  // 8 ints (256B reserved)
    int*   tok    = (int*)(ws + 256);                          // En*CAP ints (128KB)
    float* tw     = (float*)(ws + 256 + (size_t)En * CAP * 4); // En*CAP floats (128KB)

    zero_kernel<<<1024, 256, 0, stream>>>(out, Tn * Hd, counts);
    route_kernel<<<(Tn + 255) / 256, 256, 0, stream>>>(eidx, ew, counts, tok, tw);

    size_t smem = (size_t)MT * Hd * 2 + (size_t)MT * Id * 2 + MT * 4 + MT * 4; // ~288.25 KB
    (void)hipFuncSetAttribute((const void*)moe_kernel,
                              hipFuncAttributeMaxDynamicSharedMemorySize, (int)smem);
    dim3 grid(CAP / MT, En);                                   // 128 tiles x 8 experts (early-exit)
    moe_kernel<<<grid, 256, smem, stream>>>(x, gate, up, down, counts, tok, tw, out);
}
